// GnS_58918361367146
// MI455X (gfx1250) — matrix operand order repaired and checked by reading, not compiled, not run on hardware
//
#include <hip/hip_runtime.h>
#include <hip/hip_bf16.h>

#define NNODES 1048576
#define NEDGES 4194304
#define NBATCH 2048
#define GDIM   32
#define FXD    55
#define SEQ    1000
#define EMBD   128
#define LCONV  121
#define CKTOT  8000   // 1000*8
#define FCTK   3872   // 32*121
#define EPAD   136    // padded embedding row (128 data + 8 zero)

typedef __attribute__((ext_vector_type(16))) __bf16 v16bf;
typedef __attribute__((ext_vector_type(8)))  float  v8f;

union BF16x16 {
  unsigned short u[16];
  uint4 q[2];
  v16bf v;
};

__device__ __forceinline__ unsigned short f2bf(float f) {
  unsigned u = __builtin_bit_cast(unsigned, f);
  unsigned r = u + 0x7FFFu + ((u >> 16) & 1u);   // round-to-nearest-even
  return (unsigned short)(r >> 16);
}
__device__ __forceinline__ float bf2f(unsigned short h) {
  unsigned u = ((unsigned)h) << 16;
  return __builtin_bit_cast(float, u);
}

__device__ __forceinline__ v8f wmma_bf16(v16bf a, v16bf b, v8f c) {
  return __builtin_amdgcn_wmma_f32_16x16x32_bf16(false, a, false, b, (short)0, c, false, false);
}

// ---------------- weight repack: f32 [K][Nc] -> bf16 [Nc][Kpad] (transposed, zero-pad) ----
__global__ void repack_T(const float* __restrict__ in, unsigned short* __restrict__ out,
                         int K, int Nc, int Kpad) {
  int idx = blockIdx.x * 256 + threadIdx.x;
  if (idx >= Nc * Kpad) return;
  int n = idx / Kpad, k = idx - n * Kpad;
  out[idx] = (k < K) ? f2bf(in[k * Nc + n]) : (unsigned short)0;
}

// ---------------- elementwise f32 -> bf16 ------------------------------------------------
__global__ void cvt_bf16(const float* __restrict__ in, unsigned short* __restrict__ out, int n) {
  int idx = blockIdx.x * 256 + threadIdx.x;
  if (idx < n) out[idx] = f2bf(in[idx]);
}

// ---------------- layer-1 aggregation (f32, 55 dims) -------------------------------------
__global__ __launch_bounds__(256) void agg55_kernel(const int* __restrict__ ei,
                                                    const float* __restrict__ xd,
                                                    float* __restrict__ agg) {
  size_t e = (size_t)blockIdx.x * 256 + threadIdx.x;
  int s = ei[e], d = ei[NEDGES + e];
  const float* sp = xd + (size_t)s * FXD;
  float* dp = agg + (size_t)d * FXD;
  for (int c = 0; c < FXD; ++c) atomicAdd(dp + c, sp[c]);
}

// ---------------- aggregation for layers 2..5 (bf16 gather, f32 atomic) -------------------
__global__ __launch_bounds__(256) void agg32_kernel(const int* __restrict__ ei,
                                                    const unsigned short* __restrict__ x,
                                                    float* __restrict__ agg) {
  size_t e = (size_t)blockIdx.x * 256 + threadIdx.x;
  int s = ei[e], d = ei[NEDGES + e];
  const uint4* sp = (const uint4*)(x + (size_t)s * GDIM);
  BF16x16 r0, r1;
  r0.q[0] = sp[0]; r0.q[1] = sp[1];
  r1.q[0] = sp[2]; r1.q[1] = sp[3];
  float* dp = agg + (size_t)d * GDIM;
  for (int j = 0; j < 16; ++j) atomicAdd(dp + j,      bf2f(r0.u[j]));
  for (int j = 0; j < 16; ++j) atomicAdd(dp + 16 + j, bf2f(r1.u[j]));
}

// ---------------- shared GIN tail: u -> LDS -> WMMA(wb) -> relu, BN partials, pack --------
__device__ __forceinline__ void gin_tail(v8f u0, v8f u1,
    const float* __restrict__ ba, const unsigned short* __restrict__ wbT,
    const float* __restrict__ bb, unsigned short* __restrict__ hout,
    float* __restrict__ bnpart, float* lds, size_t node0, int slot) {
  int tid = threadIdx.x, wave = tid >> 5, lane = tid & 31;
  int m = lane & 15, hi = lane >> 4;
  float* tile = lds + wave * 16 * 34;
  for (int r = 0; r < 8; ++r) {
    tile[(r + hi * 8) * 34 + m]      = fmaxf(u0[r] + ba[m], 0.f);
    tile[(r + hi * 8) * 34 + 16 + m] = fmaxf(u1[r] + ba[16 + m], 0.f);
  }
  __syncthreads();
  BF16x16 a2;
  for (int j = 0; j < 8; ++j) {
    a2.u[j]     = f2bf(tile[m * 34 + hi * 16 + j]);
    a2.u[8 + j] = f2bf(tile[m * 34 + hi * 16 + 8 + j]);
  }
  __syncthreads();
  BF16x16 b0, b1;
  const unsigned short* p0 = wbT + m * 32 + hi * 16;
  const unsigned short* p1 = wbT + (16 + m) * 32 + hi * 16;
  b0.q[0] = ((const uint4*)p0)[0]; b0.q[1] = ((const uint4*)p0)[1];
  b1.q[0] = ((const uint4*)p1)[0]; b1.q[1] = ((const uint4*)p1)[1];
  v8f v0 = {}, v1 = {};
  v0 = wmma_bf16(a2.v, b0.v, v0);
  v1 = wmma_bf16(a2.v, b1.v, v1);
  float s0 = 0, q0 = 0, s1 = 0, q1 = 0;
  for (int r = 0; r < 8; ++r) {
    float a = fmaxf(v0[r] + bb[m], 0.f);
    float b = fmaxf(v1[r] + bb[16 + m], 0.f);
    tile[(r + hi * 8) * 34 + m]      = a;
    tile[(r + hi * 8) * 34 + 16 + m] = b;
    s0 += a; q0 += a * a;
    s1 += b; q1 += b * b;
  }
  s0 += __shfl_down(s0, 16); q0 += __shfl_down(q0, 16);
  s1 += __shfl_down(s1, 16); q1 += __shfl_down(q1, 16);
  if (hi == 0) {
    atomicAdd(&bnpart[slot * 64 + m],      s0);
    atomicAdd(&bnpart[slot * 64 + 32 + m], q0);
    atomicAdd(&bnpart[slot * 64 + 16 + m], s1);
    atomicAdd(&bnpart[slot * 64 + 48 + m], q1);
  }
  __syncthreads();
  BF16x16 ov;
  for (int j = 0; j < 16; ++j) ov.u[j] = f2bf(tile[m * 34 + hi * 16 + j]);
  uint4* gp = (uint4*)(hout + (node0 + m) * GDIM + hi * 16);
  gp[0] = ov.q[0]; gp[1] = ov.q[1];
}

// ---------------- GIN layer 1 MLP (55->32->32, K padded to 64) ---------------------------
// Stages (xd+agg) rows into a zero-padded LDS tile so fragment builds have no guards.
__global__ __launch_bounds__(256) void gin_mlp1(
    const float* __restrict__ xd, const float* __restrict__ agg,
    const unsigned short* __restrict__ w1aT, const float* __restrict__ b1a,
    const unsigned short* __restrict__ w1bT, const float* __restrict__ b1b,
    unsigned short* __restrict__ hout, float* __restrict__ bnpart) {
  __shared__ float xstage[128 * 64];   // 32 KB: 128 nodes x 64 (cols 55..63 stay zero)
  __shared__ float lds[8 * 16 * 34];
  int tid = threadIdx.x, wave = tid >> 5, lane = tid & 31;
  int m = lane & 15, hi = lane >> 4;
  size_t nodebase = (size_t)blockIdx.x * 128;
  // phase 1: zero the tile (covers the K pad)
  for (int i = tid; i < 128 * 64; i += 256) xstage[i] = 0.f;
  __syncthreads();
  // phase 2: coalesced fill of the 55 valid columns (27 or 28 iterations per thread)
  const float* xrow = xd + nodebase * FXD;
  const float* grow = agg + nodebase * FXD;
  for (int i = tid; i < 128 * FXD; i += 256) {
    int row = i / FXD, col = i - row * FXD;
    xstage[row * 64 + col] = xrow[i] + grow[i];
  }
  __syncthreads();
  size_t node0 = nodebase + wave * 16;
  const float* xr = xstage + (wave * 16 + m) * 64;
  v8f u0 = {}, u1 = {};
  for (int s = 0; s < 2; ++s) {
    int k0 = s * 32;
    BF16x16 af;
    for (int j = 0; j < 8; ++j) {
      af.u[j]     = f2bf(xr[k0 + hi * 16 + j]);
      af.u[8 + j] = f2bf(xr[k0 + hi * 16 + 8 + j]);
    }
    BF16x16 b0, b1;
    const unsigned short* p0 = w1aT + m * 64 + k0 + hi * 16;
    const unsigned short* p1 = w1aT + (16 + m) * 64 + k0 + hi * 16;
    b0.q[0] = ((const uint4*)p0)[0]; b0.q[1] = ((const uint4*)p0)[1];
    b1.q[0] = ((const uint4*)p1)[0]; b1.q[1] = ((const uint4*)p1)[1];
    u0 = wmma_bf16(af.v, b0.v, u0);
    u1 = wmma_bf16(af.v, b1.v, u1);
  }
  gin_tail(u0, u1, b1a, w1bT, b1b, hout, bnpart, lds, node0, blockIdx.x & 255);
}

// ---------------- GIN layers 2..5 MLP (32->32->32) ---------------------------------------
__global__ __launch_bounds__(256) void gin_mlp32(
    const unsigned short* __restrict__ x, const float* __restrict__ agg,
    const unsigned short* __restrict__ waT, const float* __restrict__ ba,
    const unsigned short* __restrict__ wbT, const float* __restrict__ bb,
    unsigned short* __restrict__ hout, float* __restrict__ bnpart) {
  __shared__ float lds[8 * 16 * 34];
  int tid = threadIdx.x, wave = tid >> 5, lane = tid & 31;
  int m = lane & 15, hi = lane >> 4;
  size_t node0 = (size_t)blockIdx.x * 128 + wave * 16;
  const unsigned short* xp = x + (node0 + m) * GDIM;
  const float* gp = agg + (node0 + m) * GDIM;
  // vectorized A-operand build: bf16 chunks + f32 chunks, add, repack
  BF16x16 xv;
  xv.q[0] = *(const uint4*)(xp + hi * 16);
  xv.q[1] = *(const uint4*)(xp + hi * 16 + 8);
  float4 g0 = *(const float4*)(gp + hi * 16);
  float4 g1 = *(const float4*)(gp + hi * 16 + 4);
  float4 g2 = *(const float4*)(gp + hi * 16 + 8);
  float4 g3 = *(const float4*)(gp + hi * 16 + 12);
  float ga[16] = {g0.x, g0.y, g0.z, g0.w, g1.x, g1.y, g1.z, g1.w,
                  g2.x, g2.y, g2.z, g2.w, g3.x, g3.y, g3.z, g3.w};
  BF16x16 af;
  for (int j = 0; j < 16; ++j) af.u[j] = f2bf(bf2f(xv.u[j]) + ga[j]);
  BF16x16 b0, b1;
  const unsigned short* p0 = waT + m * 32 + hi * 16;
  const unsigned short* p1 = waT + (16 + m) * 32 + hi * 16;
  b0.q[0] = ((const uint4*)p0)[0]; b0.q[1] = ((const uint4*)p0)[1];
  b1.q[0] = ((const uint4*)p1)[0]; b1.q[1] = ((const uint4*)p1)[1];
  v8f u0 = {}, u1 = {};
  u0 = wmma_bf16(af.v, b0.v, u0);
  u1 = wmma_bf16(af.v, b1.v, u1);
  gin_tail(u0, u1, ba, wbT, bb, hout, bnpart, lds, node0, blockIdx.x & 255);
}

// ---------------- BN finalize + apply ----------------------------------------------------
__global__ void bn_finalize(const float* __restrict__ part, const float* __restrict__ bng,
                            const float* __restrict__ bnb, int layer, float* __restrict__ ss) {
  int ch = threadIdx.x;  // 32 threads
  float s = 0, q = 0;
  for (int i = 0; i < 256; ++i) { s += part[i * 64 + ch]; q += part[i * 64 + 32 + ch]; }
  float inv = 1.0f / (float)NNODES;
  float mu = s * inv;
  float var = q * inv - mu * mu;
  float sc = bng[layer * 32 + ch] * rsqrtf(var + 1e-5f);
  ss[ch] = sc;
  ss[32 + ch] = bnb[layer * 32 + ch] - mu * sc;
}

__global__ __launch_bounds__(256) void bn_apply(const unsigned short* __restrict__ hin,
                                                const float* __restrict__ ss,
                                                unsigned short* __restrict__ xout) {
  size_t idx = (size_t)blockIdx.x * 256 + threadIdx.x;
  int ch = idx & 31;
  xout[idx] = f2bf(bf2f(hin[idx]) * ss[ch] + ss[32 + ch]);
}

// ---------------- global mean pool -------------------------------------------------------
__global__ __launch_bounds__(256) void pool_kernel(const unsigned short* __restrict__ x,
                                                   const int* __restrict__ xb,
                                                   float* __restrict__ pooled,
                                                   float* __restrict__ cnt) {
  size_t n = (size_t)blockIdx.x * 256 + threadIdx.x;
  int g = xb[n];
  const uint4* sp = (const uint4*)(x + n * GDIM);
  BF16x16 r0, r1;
  r0.q[0] = sp[0]; r0.q[1] = sp[1];
  r1.q[0] = sp[2]; r1.q[1] = sp[3];
  float* dp = pooled + (size_t)g * GDIM;
  for (int j = 0; j < 16; ++j) atomicAdd(dp + j,      bf2f(r0.u[j]));
  for (int j = 0; j < 16; ++j) atomicAdd(dp + 16 + j, bf2f(r1.u[j]));
  atomicAdd(cnt + g, 1.0f);
}

__global__ void pool_div(const float* __restrict__ pooled, const float* __restrict__ cnt,
                         unsigned short* __restrict__ out) {
  int idx = blockIdx.x * 256 + threadIdx.x;  // 2048*32
  int g = idx >> 5;
  out[idx] = f2bf(pooled[idx] / fmaxf(cnt[g], 1.0f));
}

// ---------------- generic bf16 WMMA GEMM: out = [relu](A @ Bt' + bias) -------------------
// A: [M][lda] bf16 row-major, Bt: [N][ldb] bf16 (transposed weight), out bf16 [M][ldo]
__global__ __launch_bounds__(256) void gemm_bf16(const unsigned short* __restrict__ A, int lda,
                                                 const unsigned short* __restrict__ Bt, int ldb,
                                                 const float* __restrict__ bias,
                                                 unsigned short* __restrict__ out, int ldo,
                                                 int K, int relu) {
  int tid = threadIdx.x, wave = tid >> 5, lane = tid & 31;
  int m = lane & 15, hi = lane >> 4;
  size_t row0 = (size_t)blockIdx.x * 128 + wave * 16;
  int col0 = blockIdx.y * 16;
  const unsigned short* arow = A + (row0 + m) * (size_t)lda;
  const unsigned short* brow = Bt + (size_t)(col0 + m) * ldb + hi * 16;
  v8f acc = {};
  for (int k0 = 0; k0 < K; k0 += 32) {
    BF16x16 af, bfv;
    af.q[0]  = *(const uint4*)(arow + k0 + hi * 16);
    af.q[1]  = *(const uint4*)(arow + k0 + hi * 16 + 8);
    bfv.q[0] = *(const uint4*)(brow + k0);
    bfv.q[1] = *(const uint4*)(brow + k0 + 8);
    acc = wmma_bf16(af.v, bfv.v, acc);
  }
  float bv = bias[col0 + m];
  for (int r = 0; r < 8; ++r) {
    float v = acc[r] + bv;
    if (relu) v = fmaxf(v, 0.f);
    out[(row0 + r + hi * 8) * (size_t)ldo + col0 + m] = f2bf(v);
  }
}

// ---------------- protein conv as per-graph WMMA GEMM ------------------------------------
// conv[b,o,l] = sum_{ck} W2[o][ck] * emb[tok[b, ck>>3]][l + (ck&7)]
__global__ __launch_bounds__(256) void prot_conv(const int* __restrict__ xt,
                                                 const unsigned short* __restrict__ embbf,
                                                 const unsigned short* __restrict__ W2,
                                                 const float* __restrict__ convb,
                                                 unsigned short* __restrict__ out) {
  __shared__ unsigned short embl[26 * EPAD];   // rows padded to 136, cols 128..135 zeroed
  __shared__ unsigned short tok[SEQ];
  int b = blockIdx.x;
  for (int i = threadIdx.x; i < 26 * EPAD; i += 256) {
    int row = i / EPAD, col = i - row * EPAD;
    embl[i] = (col < EMBD) ? embbf[row * EMBD + col] : (unsigned short)0;
  }
  for (int i = threadIdx.x; i < SEQ; i += 256) tok[i] = (unsigned short)xt[b * SEQ + i];
  __syncthreads();
  int tid = threadIdx.x, wave = tid >> 5, lane = tid & 31;
  int m = lane & 15, hi = lane >> 4;
  int ltile = wave;                      // 8 l-tiles cover l = 0..127 (121 valid)
  int colbase = ltile * 16 + m;          // l for this lane; col = l + k, k = 0..7
  const unsigned short* w2r0 = W2 + (size_t)m * CKTOT;
  const unsigned short* w2r1 = W2 + (size_t)(16 + m) * CKTOT;
  v8f acc0 = {}, acc1 = {};
  for (int kc = 0; kc < CKTOT / 32; ++kc) {
    int ckb = kc * 32;
    // within this fragment half, c changes only once: j<8 -> c0, j>=8 -> c0+1
    int c0 = (ckb + hi * 16) >> 3;
    const unsigned short* r0 = embl + (int)tok[c0]     * EPAD + colbase;
    const unsigned short* r1 = embl + (int)tok[c0 + 1] * EPAD + colbase;
    BF16x16 bfg;
    for (int j = 0; j < 8; ++j) { bfg.u[j] = r0[j]; bfg.u[8 + j] = r1[j]; }
    BF16x16 a0, a1;
    a0.q[0] = *(const uint4*)(w2r0 + ckb + hi * 16);
    a0.q[1] = *(const uint4*)(w2r0 + ckb + hi * 16 + 8);
    a1.q[0] = *(const uint4*)(w2r1 + ckb + hi * 16);
    a1.q[1] = *(const uint4*)(w2r1 + ckb + hi * 16 + 8);
    acc0 = wmma_bf16(a0.v, bfg.v, acc0);
    acc1 = wmma_bf16(a1.v, bfg.v, acc1);
  }
  int l = colbase;
  if (l < LCONV) {
    for (int r = 0; r < 8; ++r) {
      int o0 = r + hi * 8;
      out[(size_t)b * FCTK + o0 * LCONV + l] = f2bf(acc0[r] + convb[o0]);
      int o1 = 16 + r + hi * 8;
      out[(size_t)b * FCTK + o1 * LCONV + l] = f2bf(acc1[r] + convb[o1]);
    }
  }
}

// ---------------- final 256 -> 1 ---------------------------------------------------------
__global__ void c3_kernel(const unsigned short* __restrict__ z2, const float* __restrict__ c3w,
                          const float* __restrict__ c3b, float* __restrict__ out) {
  int bidx = blockIdx.x * 256 + threadIdx.x;
  if (bidx >= NBATCH) return;
  const unsigned short* z = z2 + (size_t)bidx * 256;
  float s = c3b[0];
  for (int j = 0; j < 256; ++j) s += bf2f(z[j]) * c3w[j];
  out[bidx] = s;
}

// =========================================================================================
extern "C" void kernel_launch(void* const* d_in, const int* in_sizes, int n_in,
                              void* d_out, int out_size, void* d_ws, size_t ws_size,
                              hipStream_t stream) {
  const float* xd     = (const float*)d_in[0];
  const int*   ei     = (const int*)d_in[1];
  const int*   xb     = (const int*)d_in[2];
  const int*   xt     = (const int*)d_in[3];
  const float* w1a    = (const float*)d_in[4];
  const float* b1a    = (const float*)d_in[5];
  const float* w1b    = (const float*)d_in[6];
  const float* b1b    = (const float*)d_in[7];
  const float* wa     = (const float*)d_in[8];
  const float* ba     = (const float*)d_in[9];
  const float* wb     = (const float*)d_in[10];
  const float* bb     = (const float*)d_in[11];
  const float* bng    = (const float*)d_in[12];
  const float* bnb    = (const float*)d_in[13];
  const float* fcd_w  = (const float*)d_in[14];
  const float* fcd_b  = (const float*)d_in[15];
  const float* emb    = (const float*)d_in[16];
  const float* conv_w = (const float*)d_in[17];
  const float* conv_b = (const float*)d_in[18];
  const float* fct_w  = (const float*)d_in[19];
  const float* fct_b  = (const float*)d_in[20];
  const float* c1w    = (const float*)d_in[21];
  const float* c1b    = (const float*)d_in[22];
  const float* c2w    = (const float*)d_in[23];
  const float* c2b    = (const float*)d_in[24];
  const float* c3w    = (const float*)d_in[25];
  const float* c3b    = (const float*)d_in[26];
  float* outf = (float*)d_out;

  char* w = (char*)d_ws;
  size_t off = 0;
  auto carve = [&](size_t bytes) -> void* {
    void* p = w + off;
    off = (off + bytes + 255) & ~(size_t)255;
    return p;
  };
  float*          aggf   = (float*)carve((size_t)NNODES * FXD * 4);   // layer1: 55 dims; layers 2-5 reuse first 32
  unsigned short* xcur   = (unsigned short*)carve((size_t)NNODES * GDIM * 2);
  unsigned short* hnext  = (unsigned short*)carve((size_t)NNODES * GDIM * 2);
  float*          bnpart = (float*)carve(256 * 64 * 4);
  float*          bnss   = (float*)carve(64 * 4);
  float*          pooled = (float*)carve((size_t)NBATCH * GDIM * 4);
  float*          cnt    = (float*)carve((size_t)NBATCH * 4);
  unsigned short* poolbf = (unsigned short*)carve((size_t)NBATCH * GDIM * 2);
  unsigned short* convo  = (unsigned short*)carve((size_t)NBATCH * FCTK * 2);
  unsigned short* xj     = (unsigned short*)carve((size_t)NBATCH * 256 * 2);
  unsigned short* z1     = (unsigned short*)carve((size_t)NBATCH * 1024 * 2);
  unsigned short* z2     = (unsigned short*)carve((size_t)NBATCH * 256 * 2);
  unsigned short* w1aT   = (unsigned short*)carve(32 * 64 * 2);
  unsigned short* w1bT   = (unsigned short*)carve(32 * 32 * 2);
  unsigned short* waT    = (unsigned short*)carve(4 * 32 * 32 * 2);
  unsigned short* wbT    = (unsigned short*)carve(4 * 32 * 32 * 2);
  unsigned short* fcdT   = (unsigned short*)carve(128 * 32 * 2);
  unsigned short* fctT   = (unsigned short*)carve((size_t)128 * FCTK * 2);
  unsigned short* c1T    = (unsigned short*)carve((size_t)1024 * 256 * 2);
  unsigned short* c2T    = (unsigned short*)carve((size_t)256 * 1024 * 2);
  unsigned short* W2     = (unsigned short*)carve((size_t)32 * CKTOT * 2);
  unsigned short* embbf  = (unsigned short*)carve(26 * 128 * 2);

  // ---- weight repacks (cheap, every call; deterministic) ----
  repack_T<<<(32 * 64 + 255) / 256, 256, 0, stream>>>(w1a, w1aT, FXD, 32, 64);
  repack_T<<<(32 * 32 + 255) / 256, 256, 0, stream>>>(w1b, w1bT, 32, 32, 32);
  for (int i = 0; i < 4; ++i) {
    repack_T<<<4, 256, 0, stream>>>(wa + i * 1024, waT + i * 1024, 32, 32, 32);
    repack_T<<<4, 256, 0, stream>>>(wb + i * 1024, wbT + i * 1024, 32, 32, 32);
  }
  repack_T<<<16, 256, 0, stream>>>(fcd_w, fcdT, 32, 128, 32);
  repack_T<<<(128 * FCTK + 255) / 256, 256, 0, stream>>>(fct_w, fctT, FCTK, 128, FCTK);
  repack_T<<<(1024 * 256 + 255) / 256, 256, 0, stream>>>(c1w, c1T, 256, 1024, 256);
  repack_T<<<(256 * 1024 + 255) / 256, 256, 0, stream>>>(c2w, c2T, 1024, 256, 1024);
  cvt_bf16<<<(26 * 128 + 255) / 256, 256, 0, stream>>>(emb, embbf, 26 * 128);
  cvt_bf16<<<(32 * CKTOT + 255) / 256, 256, 0, stream>>>(conv_w, W2, 32 * CKTOT);

  // ---- protein branch (independent of graph branch) ----
  prot_conv<<<NBATCH, 256, 0, stream>>>(xt, embbf, W2, conv_b, convo);

  // ---- GIN layer 1 ----
  hipMemsetAsync(aggf, 0, (size_t)NNODES * FXD * 4, stream);
  hipMemsetAsync(bnpart, 0, 256 * 64 * 4, stream);
  agg55_kernel<<<NEDGES / 256, 256, 0, stream>>>(ei, xd, aggf);
  gin_mlp1<<<NNODES / 128, 256, 0, stream>>>(xd, aggf, w1aT, b1a, w1bT, b1b, hnext, bnpart);
  bn_finalize<<<1, 32, 0, stream>>>(bnpart, bng, bnb, 0, bnss);
  bn_apply<<<NNODES * GDIM / 256, 256, 0, stream>>>(hnext, bnss, xcur);

  // ---- GIN layers 2..5 ----
  for (int l = 1; l < 5; ++l) {
    hipMemsetAsync(aggf, 0, (size_t)NNODES * GDIM * 4, stream);
    hipMemsetAsync(bnpart, 0, 256 * 64 * 4, stream);
    agg32_kernel<<<NEDGES / 256, 256, 0, stream>>>(ei, xcur, aggf);
    gin_mlp32<<<NNODES / 128, 256, 0, stream>>>(xcur, aggf,
        waT + (l - 1) * 1024, ba + (l - 1) * 32,
        wbT + (l - 1) * 1024, bb + (l - 1) * 32, hnext, bnpart);
    bn_finalize<<<1, 32, 0, stream>>>(bnpart, bng, bnb, l, bnss);
    bn_apply<<<NNODES * GDIM / 256, 256, 0, stream>>>(hnext, bnss, xcur);
  }

  // ---- pool + fc1_xd (relu) -> xj[:, 0:128] ----
  hipMemsetAsync(pooled, 0, (size_t)NBATCH * GDIM * 4, stream);
  hipMemsetAsync(cnt, 0, (size_t)NBATCH * 4, stream);
  pool_kernel<<<NNODES / 256, 256, 0, stream>>>(xcur, xb, pooled, cnt);
  pool_div<<<NBATCH * GDIM / 256, 256, 0, stream>>>(pooled, cnt, poolbf);
  gemm_bf16<<<dim3(NBATCH / 128, 128 / 16), 256, 0, stream>>>(poolbf, 32, fcdT, 32,
                                                              fcd_b, xj, 256, 32, 1);
  // ---- fct (no relu) -> xj[:, 128:256] ----
  gemm_bf16<<<dim3(NBATCH / 128, 128 / 16), 256, 0, stream>>>(convo, FCTK, fctT, FCTK,
                                                              fct_b, xj + 128, 256, FCTK, 0);
  // ---- classifier ----
  gemm_bf16<<<dim3(NBATCH / 128, 1024 / 16), 256, 0, stream>>>(xj, 256, c1T, 256,
                                                               c1b, z1, 1024, 256, 1);
  gemm_bf16<<<dim3(NBATCH / 128, 256 / 16), 256, 0, stream>>>(z1, 1024, c2T, 1024,
                                                              c2b, z2, 256, 1024, 1);
  c3_kernel<<<NBATCH / 256, 256, 0, stream>>>(z2, c3w, c3b, outf);

  (void)in_sizes; (void)n_in; (void)out_size; (void)ws_size;
}
